// FullAttention_23021024707032
// MI455X (gfx1250) — matrix operand order repaired and checked by reading, not compiled, not run on hardware
//
#include <hip/hip_runtime.h>

// ---------------- problem constants ----------------
#define Bv 2
#define Tv 2048
#define Dv 1024
#define Hv 16
#define Kv 64
static constexpr float INV_TAU = 0.125f;   // 1/sqrt(64)
static constexpr float NEG_INF = -1e30f;

typedef __attribute__((ext_vector_type(16))) __bf16 v16bf;
typedef __attribute__((ext_vector_type(8)))  float  v8f;

// ---------------- CDNA5 async global->LDS copy (ASYNCcnt-tracked) ----------------
#if defined(__gfx1250__) && \
    __has_builtin(__builtin_amdgcn_global_load_async_to_lds_b128) && \
    __has_builtin(__builtin_amdgcn_s_wait_asynccnt)
#define USE_ASYNC_LDS 1
#else
#define USE_ASYNC_LDS 0
#endif

#if USE_ASYNC_LDS
typedef int b128_t __attribute__((vector_size(16)));
typedef __attribute__((address_space(1))) b128_t* as1_b128;
typedef __attribute__((address_space(3))) b128_t* as3_b128;
#endif

__device__ __forceinline__ void copy16_g2l(const __bf16* g, __bf16* l) {
#if USE_ASYNC_LDS
  __builtin_amdgcn_global_load_async_to_lds_b128(
      (as1_b128)(__attribute__((address_space(1))) const void*)g,
      (as3_b128)(__attribute__((address_space(3))) void*)l, 0, 0);
#else
  *reinterpret_cast<uint4*>(l) = *reinterpret_cast<const uint4*>(g);
#endif
}
__device__ __forceinline__ void async_fence() {
#if USE_ASYNC_LDS
  __builtin_amdgcn_s_wait_asynccnt(0);
#endif
}

// ---------------- helpers ----------------
__device__ __forceinline__ __bf16 f2bf(float f) {
  unsigned u = __builtin_bit_cast(unsigned, f);
  u = (u + 0x7FFFu + ((u >> 16) & 1u)) >> 16;           // round-to-nearest-even
  return __builtin_bit_cast(__bf16, (unsigned short)u);
}
__device__ __forceinline__ float bf2f(__bf16 v) {
  unsigned u = ((unsigned)__builtin_bit_cast(unsigned short, v)) << 16;
  return __builtin_bit_cast(float, u);
}

// A fragment (16x32, MxK) from a row-major bf16 tile. Per ISA 7.12.2:
// lane L holds row M=L&15; lane-half selects K set {h*8+0..7, 16+h*8+0..7}.
__device__ __forceinline__ v16bf load_a_frag(const __bf16* base, int lda) {
  const int lane = threadIdx.x & 31;
  const __bf16* row = base + (lane & 15) * lda + ((lane >> 4) << 3);
  v16bf a;
#pragma unroll
  for (int e = 0; e < 8; ++e) { a[e] = row[e]; a[8 + e] = row[16 + e]; }
  return a;
}
// B fragment (32x16, KxN) from an N-major tile Bt[n][k] (k contiguous).
// lane L holds col N=L&15; lane-half selects K set {h*8+0..7, 16+h*8+0..7}.
__device__ __forceinline__ v16bf load_b_frag(const __bf16* base, int ldb) {
  const int lane = threadIdx.x & 31;
  const __bf16* row = base + (lane & 15) * ldb + ((lane >> 4) << 3);
  v16bf b;
#pragma unroll
  for (int e = 0; e < 8; ++e) { b[e] = row[e]; b[8 + e] = row[16 + e]; }
  return b;
}
__device__ __forceinline__ v8f wmma_bf16(v16bf a, v16bf b, v8f c) {
  return __builtin_amdgcn_wmma_f32_16x16x32_bf16(false, a, false, b, (short)0, c,
                                                 false, false);
}

// ---------------- kernel 1: f32 -> bf16 pack ----------------
__global__ void cast_bf16_kernel(const float* __restrict__ s, __bf16* __restrict__ d,
                                 int n) {
  int i = blockIdx.x * blockDim.x + threadIdx.x;
  if (i < n) d[i] = f2bf(s[i]);
}

// ---------------- kernel 2: RMSNorm -> bf16 ----------------
__global__ __launch_bounds__(256) void rmsnorm_kernel(const float* __restrict__ x,
                                                      const float* __restrict__ g,
                                                      __bf16* __restrict__ xt) {
  __shared__ float red[256];
  const int row = blockIdx.x, tid = threadIdx.x;
  const float4 v = *reinterpret_cast<const float4*>(x + (size_t)row * Dv + tid * 4);
  red[tid] = v.x * v.x + v.y * v.y + v.z * v.z + v.w * v.w;
  __syncthreads();
  for (int s = 128; s > 0; s >>= 1) {
    if (tid < s) red[tid] += red[tid + s];
    __syncthreads();
  }
  const float sc = rsqrtf(red[0] * (1.0f / Dv) + 1e-6f);
  const float4 gv = *reinterpret_cast<const float4*>(g + tid * 4);
  const size_t o = (size_t)row * Dv + tid * 4;
  xt[o + 0] = f2bf(v.x * sc * gv.x);
  xt[o + 1] = f2bf(v.y * sc * gv.y);
  xt[o + 2] = f2bf(v.z * sc * gv.z);
  xt[o + 3] = f2bf(v.w * sc * gv.w);
}

// ---------------- kernel 3: QKV projection GEMM ----------------
// C[m,n] = sum_d xt[m,d]*W[d,n];  m over B*T, n over H*K.  z selects W.
// z==0 writes qu = C/tau + xl_u[n] and qv = C/tau + xl_v[n].
__global__ __launch_bounds__(256) void qkv_gemm_kernel(
    const __bf16* __restrict__ xt, const __bf16* __restrict__ wq,
    const __bf16* __restrict__ wk, const __bf16* __restrict__ wv,
    const float* __restrict__ xl_u, const float* __restrict__ xl_v,
    __bf16* __restrict__ qu, __bf16* __restrict__ qv, __bf16* __restrict__ kk,
    __bf16* __restrict__ vv) {
  __shared__ alignas(16) __bf16 As[128][40];   // [m][k]
  __shared__ alignas(16) __bf16 Bt[64][40];    // [n][k] (transposed for B frags)
  const int tid = threadIdx.x, lane = tid & 31, wave = tid >> 5;
  const int m0 = blockIdx.y * 128, n0 = blockIdx.x * 64;
  const int which = blockIdx.z;
  const __bf16* w = (which == 0) ? wq : ((which == 1) ? wk : wv);
  const int wr = wave >> 1, wc = wave & 1;

  v8f acc[2][2];
#pragma unroll
  for (int i = 0; i < 2; ++i)
#pragma unroll
    for (int j = 0; j < 2; ++j) acc[i][j] = (v8f){0, 0, 0, 0, 0, 0, 0, 0};

  for (int kt = 0; kt < Dv / 32; ++kt) {
    const int kd0 = kt * 32;
    __syncthreads();
#pragma unroll
    for (int i = 0; i < 2; ++i) {          // A: 128x32 = 512 chunks of 8 bf16
      int c = tid + i * 256;
      int row = c >> 2, col = (c & 3) << 3;
      copy16_g2l(xt + (size_t)(m0 + row) * Dv + kd0 + col, &As[row][col]);
      if (i == 0 && kt + 1 < Dv / 32)      // warm L2 for next K-slice
        __builtin_prefetch(xt + (size_t)(m0 + row) * Dv + kd0 + 32 + col, 0, 0);
    }
    {                                       // B: 32x64, stored transposed
      int kdl = tid >> 3, nb = (tid & 7) << 3;
      union { uint4 u; __bf16 h[8]; } t;
      t.u = *reinterpret_cast<const uint4*>(w + (size_t)(kd0 + kdl) * (Hv * Kv) + n0 + nb);
#pragma unroll
      for (int e = 0; e < 8; ++e) Bt[nb + e][kdl] = t.h[e];
    }
    async_fence();
    __syncthreads();
    v16bf a0 = load_a_frag(&As[wr * 32][0], 40);
    v16bf a1 = load_a_frag(&As[wr * 32 + 16][0], 40);
    v16bf b0 = load_b_frag(&Bt[wc * 32][0], 40);
    v16bf b1 = load_b_frag(&Bt[wc * 32 + 16][0], 40);
    acc[0][0] = wmma_bf16(a0, b0, acc[0][0]);
    acc[0][1] = wmma_bf16(a0, b1, acc[0][1]);
    acc[1][0] = wmma_bf16(a1, b0, acc[1][0]);
    acc[1][1] = wmma_bf16(a1, b1, acc[1][1]);
  }

  // hoist the per-column XL biases (2 columns per thread)
  float xu[2] = {0.f, 0.f}, xv[2] = {0.f, 0.f};
  if (which == 0) {
#pragma unroll
    for (int tj = 0; tj < 2; ++tj) {
      int n = n0 + wc * 32 + tj * 16 + (lane & 15);
      xu[tj] = xl_u[n];
      xv[tj] = xl_v[n];
    }
  }
#pragma unroll
  for (int ti = 0; ti < 2; ++ti)
#pragma unroll
    for (int tj = 0; tj < 2; ++tj)
#pragma unroll
      for (int r = 0; r < 8; ++r) {
        int m = m0 + wr * 32 + ti * 16 + ((lane >> 4) << 3) + r;
        int n = n0 + wc * 32 + tj * 16 + (lane & 15);
        int bb = m >> 11, t = m & (Tv - 1);
        int h = n >> 6, c = n & 63;
        size_t o = (((size_t)(bb * Hv + h) * Tv + t) << 6) + c;
        float val = acc[ti][tj][r];
        if (which == 0) {
          qu[o] = f2bf(val * INV_TAU + xu[tj]);
          qv[o] = f2bf(val * INV_TAU + xv[tj]);
        } else if (which == 1) {
          kk[o] = f2bf(val);
        } else {
          vv[o] = f2bf(val);
        }
      }
}

// ---------------- kernel 4: causal flash attention with XL bias ----------------
// grid = (T/64, B*H); 128 threads (4 waves), each wave owns 16 query rows.
__global__ __launch_bounds__(128) void flash_attn_kernel(
    const __bf16* __restrict__ qu, const __bf16* __restrict__ qv,
    const __bf16* __restrict__ kk, const __bf16* __restrict__ vv,
    const __bf16* __restrict__ rr, __bf16* __restrict__ obuf) {
  __shared__ alignas(16) __bf16 Kt[64][72];          // K tile row-major [j][k]
  __shared__ alignas(16) __bf16 Vt[64][72];          // V transposed [vdim][j]
  __shared__ alignas(16) __bf16 Rw[128][72];         // xl_r window [c][k]
  __shared__ alignas(16) __bf16 Gs[4][16][128];      // BD window spill, per wave
  __shared__ alignas(16) __bf16 Ps[4][16][72];       // P transpose buffer, per wave

  const int tid = threadIdx.x, lane = tid & 31, wave = tid >> 5;
  const int qt = blockIdx.x, bh = blockIdx.y;
  const int b = bh >> 4, h = bh & 15;
  const int i0 = qt * 64;
  const size_t headbase = (size_t)(b * Hv + h) * Tv * Kv;
  const int Mh = (lane >> 4) << 3;                   // C-layout row base for this half

  // Q fragments straight from global (A layout: one row per lane)
  const __bf16* qur = qu + headbase + (size_t)(i0 + wave * 16 + (lane & 15)) * 64;
  const __bf16* qvr = qv + headbase + (size_t)(i0 + wave * 16 + (lane & 15)) * 64;
  const int hb = (lane >> 4) << 3;
  v16bf au0, au1, av0, av1;
#pragma unroll
  for (int e = 0; e < 8; ++e) {
    au0[e] = qur[hb + e];      au0[8 + e] = qur[16 + hb + e];
    au1[e] = qur[32 + hb + e]; au1[8 + e] = qur[48 + hb + e];
    av0[e] = qvr[hb + e];      av0[8 + e] = qvr[16 + hb + e];
    av1[e] = qvr[32 + hb + e]; av1[8 + e] = qvr[48 + hb + e];
  }

  v8f Oacc[4];
#pragma unroll
  for (int vt = 0; vt < 4; ++vt) Oacc[vt] = (v8f){0, 0, 0, 0, 0, 0, 0, 0};
  float mrun[8], lrun[8];
#pragma unroll
  for (int r = 0; r < 8; ++r) { mrun[r] = NEG_INF; lrun[r] = 0.0f; }

  for (int jt = 0; jt <= qt; ++jt) {
    const int j0 = jt * 64;
    __syncthreads();
#pragma unroll
    for (int i = 0; i < 4; ++i) {          // K tile (row-major == B-frag layout)
      int c = tid + i * 128;
      int row = c >> 3, col = (c & 7) << 3;
      copy16_g2l(kk + headbase + (size_t)(j0 + row) * 64 + col, &Kt[row][col]);
    }
#pragma unroll
    for (int i = 0; i < 4; ++i) {          // V tile transposed (manual path)
      int c = tid + i * 128;
      int row = c >> 3, col = (c & 7) << 3;
      union { uint4 u; __bf16 hh[8]; } t;
      t.u = *reinterpret_cast<const uint4*>(vv + headbase + (size_t)(j0 + row) * 64 + col);
#pragma unroll
      for (int e = 0; e < 8; ++e) Vt[col + e][row] = t.hh[e];
    }
    const int w0 = Tv - 64 - i0 + j0;      // window start: bd row = w0 + (63-n+m)
#pragma unroll
    for (int i = 0; i < 8; ++i) {          // xl_r window (128 rows)
      int c = tid + i * 128;
      int row = c >> 3, col = (c & 7) << 3;
      int rrow = w0 + row; if (rrow > Tv - 1) rrow = Tv - 1;
      copy16_g2l(rr + ((size_t)h * Tv + rrow) * 64 + col, &Rw[row][col]);
    }
    async_fence();
    __syncthreads();

    // ----- AC scores: S = Qu * K^T (16x64 per wave) -----
    float s[4][8];
#pragma unroll
    for (int nt = 0; nt < 4; ++nt) {
      v8f sa = (v8f){0, 0, 0, 0, 0, 0, 0, 0};
      sa = wmma_bf16(au0, load_b_frag(&Kt[nt * 16][0], 72), sa);
      sa = wmma_bf16(au1, load_b_frag(&Kt[nt * 16][32], 72), sa);
#pragma unroll
      for (int r = 0; r < 8; ++r) s[nt][r] = sa[r];
    }
    // ----- BD window GEMM: G = Qv * Rw^T (16x128), spill to LDS -----
#pragma unroll
    for (int ct = 0; ct < 8; ++ct) {
      v8f g = (v8f){0, 0, 0, 0, 0, 0, 0, 0};
      g = wmma_bf16(av0, load_b_frag(&Rw[ct * 16][0], 72), g);
      g = wmma_bf16(av1, load_b_frag(&Rw[ct * 16][32], 72), g);
#pragma unroll
      for (int r = 0; r < 8; ++r)
        Gs[wave][Mh + r][ct * 16 + (lane & 15)] = f2bf(g[r]);
    }
    // ----- combine: S += band(G), causal mask on diagonal block -----
#pragma unroll
    for (int nt = 0; nt < 4; ++nt)
#pragma unroll
      for (int r = 0; r < 8; ++r) {
        int M = Mh + r;                        // row in wave strip
        int n = wave * 16 + M;                 // row in 64-query block
        int m = nt * 16 + (lane & 15);         // col in 64-key block
        float sv = s[nt][r] + bf2f(Gs[wave][M][63 - n + m]);
        if (jt == qt && (j0 + m) > (i0 + n)) sv = NEG_INF;
        s[nt][r] = sv;
      }
    // ----- online softmax (row = one lane-half) -----
#pragma unroll
    for (int r = 0; r < 8; ++r) {
      float mx = fmaxf(fmaxf(s[0][r], s[1][r]), fmaxf(s[2][r], s[3][r]));
#pragma unroll
      for (int sh = 1; sh < 16; sh <<= 1) mx = fmaxf(mx, __shfl_xor(mx, sh, 32));
      float nm = fmaxf(mrun[r], mx);
      float sc = __expf(mrun[r] - nm);
      mrun[r] = nm;
      float rs = 0.0f;
#pragma unroll
      for (int nt = 0; nt < 4; ++nt) {
        float p = __expf(s[nt][r] - nm);
        Ps[wave][Mh + r][nt * 16 + (lane & 15)] = f2bf(p);
        rs += p;
      }
#pragma unroll
      for (int sh = 1; sh < 16; sh <<= 1) rs += __shfl_xor(rs, sh, 32);
      lrun[r] = lrun[r] * sc + rs;
#pragma unroll
      for (int vt = 0; vt < 4; ++vt) Oacc[vt][r] *= sc;
    }
    // ----- PV: O += P * V (P re-read as A fragments from LDS) -----
    v16bf pa0 = load_a_frag(&Ps[wave][0][0], 72);
    v16bf pa1 = load_a_frag(&Ps[wave][0][32], 72);
#pragma unroll
    for (int vt = 0; vt < 4; ++vt) {
      Oacc[vt] = wmma_bf16(pa0, load_b_frag(&Vt[vt * 16][0], 72), Oacc[vt]);
      Oacc[vt] = wmma_bf16(pa1, load_b_frag(&Vt[vt * 16][32], 72), Oacc[vt]);
    }
  }

  // epilogue: O/l -> bf16 [B*T, H*K]
#pragma unroll
  for (int vt = 0; vt < 4; ++vt)
#pragma unroll
    for (int r = 0; r < 8; ++r) {
      int i = i0 + wave * 16 + Mh + r;
      int col = h * 64 + vt * 16 + (lane & 15);
      obuf[((size_t)(b * Tv + i) << 10) + col] = f2bf(Oacc[vt][r] / lrun[r]);
    }
}

// ---------------- kernel 5: output projection GEMM (bf16 x bf16 -> f32) ----------------
__global__ __launch_bounds__(256) void out_gemm_kernel(const __bf16* __restrict__ ob,
                                                       const __bf16* __restrict__ wr,
                                                       float* __restrict__ out) {
  __shared__ alignas(16) __bf16 As[128][40];
  __shared__ alignas(16) __bf16 Bt[64][40];
  const int tid = threadIdx.x, lane = tid & 31, wave = tid >> 5;
  const int m0 = blockIdx.y * 128, n0 = blockIdx.x * 64;
  const int wrw = wave >> 1, wc = wave & 1;

  v8f acc[2][2];
#pragma unroll
  for (int i = 0; i < 2; ++i)
#pragma unroll
    for (int j = 0; j < 2; ++j) acc[i][j] = (v8f){0, 0, 0, 0, 0, 0, 0, 0};

  for (int kt = 0; kt < (Hv * Kv) / 32; ++kt) {
    const int kd0 = kt * 32;
    __syncthreads();
#pragma unroll
    for (int i = 0; i < 2; ++i) {
      int c = tid + i * 256;
      int row = c >> 2, col = (c & 3) << 3;
      copy16_g2l(ob + (size_t)(m0 + row) * (Hv * Kv) + kd0 + col, &As[row][col]);
      if (i == 0 && kt + 1 < (Hv * Kv) / 32)
        __builtin_prefetch(ob + (size_t)(m0 + row) * (Hv * Kv) + kd0 + 32 + col, 0, 0);
    }
    {
      int kdl = tid >> 3, nb = (tid & 7) << 3;
      union { uint4 u; __bf16 h[8]; } t;
      t.u = *reinterpret_cast<const uint4*>(wr + (size_t)(kd0 + kdl) * Dv + n0 + nb);
#pragma unroll
      for (int e = 0; e < 8; ++e) Bt[nb + e][kdl] = t.h[e];
    }
    async_fence();
    __syncthreads();
    v16bf a0 = load_a_frag(&As[wrw * 32][0], 40);
    v16bf a1 = load_a_frag(&As[wrw * 32 + 16][0], 40);
    v16bf b0 = load_b_frag(&Bt[wc * 32][0], 40);
    v16bf b1 = load_b_frag(&Bt[wc * 32 + 16][0], 40);
    acc[0][0] = wmma_bf16(a0, b0, acc[0][0]);
    acc[0][1] = wmma_bf16(a0, b1, acc[0][1]);
    acc[1][0] = wmma_bf16(a1, b0, acc[1][0]);
    acc[1][1] = wmma_bf16(a1, b1, acc[1][1]);
  }
#pragma unroll
  for (int ti = 0; ti < 2; ++ti)
#pragma unroll
    for (int tj = 0; tj < 2; ++tj)
#pragma unroll
      for (int r = 0; r < 8; ++r) {
        int m = m0 + wrw * 32 + ti * 16 + ((lane >> 4) << 3) + r;
        int n = n0 + wc * 32 + tj * 16 + (lane & 15);
        out[(size_t)m * Dv + n] = acc[ti][tj][r];
      }
}

// ---------------- launch ----------------
extern "C" void kernel_launch(void* const* d_in, const int* in_sizes, int n_in,
                              void* d_out, int out_size, void* d_ws, size_t ws_size,
                              hipStream_t stream) {
  (void)in_sizes; (void)n_in; (void)out_size; (void)ws_size;
  const float* x     = (const float*)d_in[0];
  const float* g     = (const float*)d_in[1];
  const float* w_q   = (const float*)d_in[2];
  const float* w_k   = (const float*)d_in[3];
  const float* w_v   = (const float*)d_in[4];
  const float* w_res = (const float*)d_in[5];
  const float* xl_r  = (const float*)d_in[6];
  const float* xl_u  = (const float*)d_in[7];
  const float* xl_v  = (const float*)d_in[8];
  float* out = (float*)d_out;

  // workspace carve-up (~60 MB of bf16)
  const size_t NROW = (size_t)Bv * Tv;                 // 4096
  const size_t WSZ  = (size_t)Dv * Hv * Kv;            // 1,048,576
  const size_t HEAD = (size_t)Bv * Hv * Tv * Kv;       // 4,194,304
  __bf16* p   = (__bf16*)d_ws;
  __bf16* xt  = p; p += NROW * Dv;
  __bf16* wqb = p; p += WSZ;
  __bf16* wkb = p; p += WSZ;
  __bf16* wvb = p; p += WSZ;
  __bf16* wrb = p; p += WSZ;
  __bf16* rrb = p; p += (size_t)Hv * Tv * Kv;
  __bf16* qub = p; p += HEAD;
  __bf16* qvb = p; p += HEAD;
  __bf16* kkb = p; p += HEAD;
  __bf16* vvb = p; p += HEAD;
  __bf16* obf = p; p += NROW * Dv;

  const int nW = (int)WSZ, nR = Hv * Tv * Kv;
  cast_bf16_kernel<<<(nW + 255) / 256, 256, 0, stream>>>(w_q, wqb, nW);
  cast_bf16_kernel<<<(nW + 255) / 256, 256, 0, stream>>>(w_k, wkb, nW);
  cast_bf16_kernel<<<(nW + 255) / 256, 256, 0, stream>>>(w_v, wvb, nW);
  cast_bf16_kernel<<<(nW + 255) / 256, 256, 0, stream>>>(w_res, wrb, nW);
  cast_bf16_kernel<<<(nR + 255) / 256, 256, 0, stream>>>(xl_r, rrb, nR);

  rmsnorm_kernel<<<(int)NROW, 256, 0, stream>>>(x, g, xt);

  dim3 gq((Hv * Kv) / 64, (int)NROW / 128, 3);
  qkv_gemm_kernel<<<gq, 256, 0, stream>>>(xt, wqb, wkb, wvb, xl_u, xl_v,
                                          qub, qvb, kkb, vvb);

  dim3 gf(Tv / 64, Bv * Hv);
  flash_attn_kernel<<<gf, 128, 0, stream>>>(qub, qvb, kkb, vvb, rrb, obf);

  dim3 go(Dv / 64, (int)NROW / 128);
  out_gemm_kernel<<<go, 256, 0, stream>>>(obf, wrb, out);
}
